// Transformer_23794118820558
// MI455X (gfx1250) — matrix operand order repaired and checked by reading, not compiled, not run on hardware
//
#include <hip/hip_runtime.h>
#include <math.h>
#include <stdint.h>

// ---------------- model dims ----------------
static constexpr int Bb  = 2;
static constexpr int Ss  = 2048;
static constexpr int Dd  = 512;
static constexpr int Hh  = 8;
static constexpr int Kk  = 64;
static constexpr int Ff  = 2048;
static constexpr int Ll  = 4;
static constexpr int Mm  = Bb * Ss;    // 4096 activation rows
static constexpr int HKk = Hh * Kk;    // 512

typedef __attribute__((ext_vector_type(16))) __bf16 v16bf;
typedef __attribute__((ext_vector_type(8)))  float  v8f;
typedef __attribute__((ext_vector_type(4)))  int    v4i;

#define DEV __device__ __forceinline__

// async global->LDS copy path (gfx1250) if the toolchain exposes it
#if defined(__has_builtin)
#if __has_builtin(__builtin_amdgcn_global_load_async_to_lds_b128) && \
    __has_builtin(__builtin_amdgcn_s_wait_asynccnt)
#define USE_ASYNC 1
#endif
#endif
#ifndef USE_ASYNC
#define USE_ASYNC 0
#endif

DEV unsigned short f2bf(float f) {
  unsigned int u = __float_as_uint(f);
  u += 0x7FFFu + ((u >> 16) & 1u);   // round to nearest even
  return (unsigned short)(u >> 16);
}

union ABFrag { v16bf v; uint4 q[2]; unsigned short s[16]; };
union CFrag  { v8f  v; float f[8]; };

DEV v8f wmma_bf16(const ABFrag& a, const ABFrag& b, v8f c) {
  return __builtin_amdgcn_wmma_f32_16x16x32_bf16(false, a.v, false, b.v,
                                                 (short)0, c, false, false);
}

// copy 16 bytes global -> LDS
DEV void cp16(const unsigned short* g, unsigned short* l) {
#if USE_ASYNC
  typedef __attribute__((address_space(1))) v4i* gp_t;
  typedef __attribute__((address_space(3))) v4i* lp_t;
  __builtin_amdgcn_global_load_async_to_lds_b128(
      (gp_t)(unsigned long long)(uintptr_t)g,
      (lp_t)(unsigned int)(uintptr_t)l, 0, 0);
#else
  *(uint4*)l = *(const uint4*)g;
#endif
}
DEV void cp_wait() {
#if USE_ASYNC
  __builtin_amdgcn_s_wait_asynccnt(0);
#endif
}

// ---------------- embedding + positional ----------------
__global__ void embed_k(const int* __restrict__ x, const float* __restrict__ emb,
                        const float* __restrict__ pos, float* __restrict__ hf, int n) {
  int i = blockIdx.x * blockDim.x + threadIdx.x;
  if (i >= n) return;
  int row = i >> 9;          // / Dd
  int d   = i & (Dd - 1);
  int s   = row & (Ss - 1);  // row = b*Ss + s
  int tok = x[row];
  hf[i] = emb[(size_t)tok * Dd + d] + pos[(size_t)s * Dd + d];
}

// ---------------- f32 -> bf16 convert ----------------
__global__ void conv_k(const float* __restrict__ in, unsigned short* __restrict__ out, int n) {
  int i = blockIdx.x * blockDim.x + threadIdx.x;
  if (i < n) out[i] = f2bf(in[i]);
}

// ---------------- transpose+convert: W[R][C] f32 -> Wt[C][R] bf16 ----------------
__global__ void trans_k(const float* __restrict__ W, unsigned short* __restrict__ Wt,
                        int R, int C) {
  int i = blockIdx.x * blockDim.x + threadIdx.x;
  if (i >= R * C) return;
  int r = i / C, c = i - r * C;
  Wt[(size_t)c * R + r] = f2bf(W[i]);
}

// ---------------- V head-transpose: [b,s,h,c](bf16) -> Vt[b,h,c,s](bf16) --------
__global__ void vtrans_k(const unsigned short* __restrict__ V,
                         unsigned short* __restrict__ Vt, int n) {
  int i = blockIdx.x * blockDim.x + threadIdx.x;
  if (i >= n) return;
  int row = i >> 9;          // b*Ss + s
  int col = i & (HKk - 1);   // h*64 + c
  int h = col >> 6, c = col & (Kk - 1);
  int b = row >> 11, s = row & (Ss - 1);
  Vt[(((size_t)(b * Hh + h)) * Kk + c) * Ss + s] = V[i];
}

// ---------------- GEMM: out = A[M,Kd](bf16) x Wt[N,Kd]^T + bias ----------------
// Block: 256 threads (8 waves).  Block tile 128x128, K chunks of 32 staged
// through double-buffered LDS (async DMA copies).  Each wave owns a 32x64
// sub-tile (2x4 WMMA fragments).
// MODE 0: bf16 out * omul; MODE 1: bf16 out + ReLU;
// MODE 3: f32 residual accumulate + bf16 dual write.
static constexpr int LSTR = 40;       // padded LDS row stride (elements)

template <int MODE>
__global__ __launch_bounds__(256, 1)
void gemm_k(const unsigned short* __restrict__ A,
            const unsigned short* __restrict__ Bt,
            const float* __restrict__ bias,
            unsigned short* __restrict__ Oh,
            float* __restrict__ Of,
            int M, int N, int Kd, float omul) {
  __shared__ unsigned short Als[2][128 * LSTR];
  __shared__ unsigned short Bls[2][128 * LSTR];

  const int tid  = threadIdx.x;
  const int lane = tid & 31;
  const int wv   = tid >> 5;
  const int nl = lane & 15, kg = lane >> 4;
  const int wm0 = (wv & 3) * 32;       // wave row offset in block tile
  const int wn0 = (wv >> 2) * 64;      // wave col offset in block tile
  const int m0 = blockIdx.y * 128, n0 = blockIdx.x * 128;

  // staging coords: each thread copies 32B of A-tile and 32B of B-tile
  const int srow  = tid >> 1;          // 0..127
  const int shalf = (tid & 1) * 16;    // element offset within the 32-elem chunk

  CFrag acc[2][4];
#pragma unroll
  for (int a = 0; a < 2; ++a)
#pragma unroll
    for (int b = 0; b < 4; ++b)
#pragma unroll
      for (int i = 0; i < 8; ++i) acc[a][b].f[i] = 0.0f;

  // stage first K chunk
  {
    const unsigned short* ga = A + (size_t)(m0 + srow) * Kd + shalf;
    const unsigned short* gb = Bt + (size_t)(n0 + srow) * Kd + shalf;
    unsigned short* la = &Als[0][srow * LSTR + shalf];
    unsigned short* lb = &Bls[0][srow * LSTR + shalf];
    cp16(ga, la); cp16(ga + 8, la + 8);
    cp16(gb, lb); cp16(gb + 8, lb + 8);
  }
  cp_wait();
  __syncthreads();

  int pp = 0;
  for (int kc = 0; kc < Kd; kc += 32) {
    const int nk = kc + 32;
    if (nk < Kd) {                     // stage next chunk into other buffer
      const unsigned short* ga = A + (size_t)(m0 + srow) * Kd + nk + shalf;
      const unsigned short* gb = Bt + (size_t)(n0 + srow) * Kd + nk + shalf;
      unsigned short* la = &Als[pp ^ 1][srow * LSTR + shalf];
      unsigned short* lb = &Bls[pp ^ 1][srow * LSTR + shalf];
      cp16(ga, la); cp16(ga + 8, la + 8);
      cp16(gb, lb); cp16(gb + 8, lb + 8);
    }

    // fragments from LDS buffer pp
    ABFrag af[2], bfr[4];
#pragma unroll
    for (int sm = 0; sm < 2; ++sm) {
      const unsigned short* p = &Als[pp][(wm0 + sm * 16 + nl) * LSTR + kg * 8];
      af[sm].q[0] = *(const uint4*)p;         // K = kg*8 + 0..7
      af[sm].q[1] = *(const uint4*)(p + 16);  // K = 16 + kg*8 + 0..7
    }
#pragma unroll
    for (int sn = 0; sn < 4; ++sn) {
      const unsigned short* p = &Bls[pp][(wn0 + sn * 16 + nl) * LSTR + kg * 8];
      bfr[sn].q[0] = *(const uint4*)p;        // K = kg*8 + 0..7
      bfr[sn].q[1] = *(const uint4*)(p + 16); // K = 16 + kg*8 + 0..7
    }
#pragma unroll
    for (int sm = 0; sm < 2; ++sm)
#pragma unroll
      for (int sn = 0; sn < 4; ++sn)
        acc[sm][sn].v = wmma_bf16(af[sm], bfr[sn], acc[sm][sn].v);

    if (nk < Kd) {
      cp_wait();
      __syncthreads();
    }
    pp ^= 1;
  }

#pragma unroll
  for (int sm = 0; sm < 2; ++sm)
#pragma unroll
    for (int sn = 0; sn < 4; ++sn) {
      int col = n0 + wn0 + sn * 16 + nl;
      float bv = bias[col];
#pragma unroll
      for (int i = 0; i < 8; ++i) {
        int row = m0 + wm0 + sm * 16 + kg * 8 + i;
        float v = acc[sm][sn].f[i] + bv;
        size_t idx = (size_t)row * N + col;
        if (MODE == 0) {
          Oh[idx] = f2bf(v * omul);
        } else if (MODE == 1) {
          Oh[idx] = f2bf(v > 0.0f ? v : 0.0f);
        } else {                        // MODE 3: residual + bf16 dual write
          float r = Of[idx] + v;
          Of[idx] = r;
          Oh[idx] = f2bf(r);
        }
      }
    }
}

// ---------------- attention chunk (32 keys) ----------------
// MASKED=false: chunk fully below the causal diagonal (no compares needed).
template <bool MASKED>
DEV void attn_chunk(int kb0, int qt, int nl, int kg,
                    const unsigned short* __restrict__ krow0,  // K base + headoff
                    const unsigned short* __restrict__ vt,     // head-major V
                    const ABFrag qa[2], CFrag o[4],
                    float* mstat, float* lstat, unsigned short* Pl) {
  // S = Q * K^T for two 16-key subtiles (Q already scaled)
  CFrag sf[2];
#pragma unroll
  for (int t = 0; t < 2; ++t) {
#pragma unroll
    for (int i = 0; i < 8; ++i) sf[t].f[i] = 0.0f;
#pragma unroll
    for (int kc2 = 0; kc2 < 2; ++kc2) {
      ABFrag kf;
      const uint4* p = (const uint4*)(krow0 + (size_t)(kb0 + t * 16 + nl) * HKk +
                                      kc2 * 32 + kg * 8);
      kf.q[0] = p[0];
      kf.q[1] = p[2];
      sf[t].v = wmma_bf16(qa[kc2], kf, sf[t].v);
    }
    if (MASKED) {
      int keyc = kb0 + t * 16 + nl;
#pragma unroll
      for (int i = 0; i < 8; ++i) {
        int qr = qt * 16 + kg * 8 + i;
        sf[t].f[i] = (keyc <= qr) ? sf[t].f[i] : -__builtin_inff();
      }
    }
  }

  // online softmax row stats (row r lives across the 16 lanes of one half)
  float fv[8];
#pragma unroll
  for (int i = 0; i < 8; ++i) {
    float mx = fmaxf(sf[0].f[i], sf[1].f[i]);
    mx = fmaxf(mx, __shfl_xor(mx, 1, 32));
    mx = fmaxf(mx, __shfl_xor(mx, 2, 32));
    mx = fmaxf(mx, __shfl_xor(mx, 4, 32));
    mx = fmaxf(mx, __shfl_xor(mx, 8, 32));
    float mn = fmaxf(mstat[i], mx);
    float p0 = __expf(sf[0].f[i] - mn);
    float p1 = __expf(sf[1].f[i] - mn);
    sf[0].f[i] = p0; sf[1].f[i] = p1;
    float sm = p0 + p1;
    sm += __shfl_xor(sm, 1, 32);
    sm += __shfl_xor(sm, 2, 32);
    sm += __shfl_xor(sm, 4, 32);
    sm += __shfl_xor(sm, 8, 32);
    float fexp = __expf(mstat[i] - mn);
    lstat[i] = lstat[i] * fexp + sm;
    mstat[i] = mn;
    fv[i] = fexp;
  }
#pragma unroll
  for (int t = 0; t < 4; ++t)
#pragma unroll
    for (int i = 0; i < 8; ++i) o[t].f[i] *= fv[i];

  // write P to LDS (C-layout -> row-major 16x32 tile)
#pragma unroll
  for (int i = 0; i < 8; ++i) {
    int r = kg * 8 + i;
    Pl[r * 32 + nl]      = f2bf(sf[0].f[i]);
    Pl[r * 32 + 16 + nl] = f2bf(sf[1].f[i]);
  }
  __syncthreads();

  // read P as A-fragment
  ABFrag pa;
  pa.q[0] = *(const uint4*)(&Pl[nl * 32 + kg * 8]);
  pa.q[1] = *(const uint4*)(&Pl[nl * 32 + 16 + kg * 8]);

  // O += P * V : B-fragments straight from head-major Vt (contiguous in keys)
#pragma unroll
  for (int t4 = 0; t4 < 4; ++t4) {
    ABFrag vf;
    const uint4* p = (const uint4*)(vt + (size_t)(t4 * 16 + nl) * Ss + kb0 + kg * 8);
    vf.q[0] = p[0];
    vf.q[1] = p[2];
    o[t4].v = wmma_bf16(pa, vf, o[t4].v);
  }
  __syncthreads();
}

// ---------------- flash attention: per (b, h, 16-query tile) ----------------
// Q pre-scaled by 1/sqrt(K).  V supplied head-major transposed: Vt[b,h,c,s].
__global__ __launch_bounds__(32, 1)
void attn_k(const unsigned short* __restrict__ Q,
            const unsigned short* __restrict__ Kb,
            const unsigned short* __restrict__ Vt,
            unsigned short* __restrict__ Ob) {
  __shared__ unsigned short Pl[16 * 32];   // P tile, row-major 16x32 bf16

  const int lane = threadIdx.x;
  const int nl = lane & 15, kg = lane >> 4;
  const int nqt = Ss / 16;
  const int qt = blockIdx.x % nqt;
  const int bh = blockIdx.x / nqt;
  const int h  = bh % Hh, b = bh / Hh;
  const size_t headoff = (size_t)h * Kk;
  const unsigned short* krow0 = Kb + (size_t)(b * Ss) * HKk + headoff;
  const unsigned short* vt = Vt + ((size_t)(b * Hh + h) * Kk) * Ss;
  const float NEG = -__builtin_inff();

  // Q fragments (A-layout), two K-dim chunks of 32
  ABFrag qa[2];
#pragma unroll
  for (int kc2 = 0; kc2 < 2; ++kc2) {
    const uint4* p = (const uint4*)(Q + (size_t)(b * Ss + qt * 16 + nl) * HKk +
                                    headoff + kc2 * 32 + kg * 8);
    qa[kc2].q[0] = p[0];
    qa[kc2].q[1] = p[2];
  }

  CFrag o[4];
  float mstat[8], lstat[8];
#pragma unroll
  for (int t = 0; t < 4; ++t)
#pragma unroll
    for (int i = 0; i < 8; ++i) o[t].f[i] = 0.0f;
#pragma unroll
  for (int i = 0; i < 8; ++i) { mstat[i] = NEG; lstat[i] = 0.0f; }

  // chunks strictly below the diagonal need no mask; exactly one (the
  // diagonal chunk) needs the causal compare.
  const int nfull = qt >> 1;                 // (qt*16)/32
  for (int ch = 0; ch < nfull; ++ch)
    attn_chunk<false>(ch * 32, qt, nl, kg, krow0, vt, qa, o, mstat, lstat, Pl);
  attn_chunk<true>(nfull * 32, qt, nl, kg, krow0, vt, qa, o, mstat, lstat, Pl);

  // normalize and store bf16 output at [b, s, h*64 + k]
  float rl[8];
#pragma unroll
  for (int i = 0; i < 8; ++i) rl[i] = 1.0f / lstat[i];
#pragma unroll
  for (int t4 = 0; t4 < 4; ++t4) {
    size_t col = headoff + t4 * 16 + nl;
#pragma unroll
    for (int i = 0; i < 8; ++i) {
      size_t row = (size_t)(b * Ss + qt * 16 + kg * 8 + i);
      Ob[row * HKk + col] = f2bf(o[t4].f[i] * rl[i]);
    }
  }
}

// ---------------- host orchestration ----------------
extern "C" void kernel_launch(void* const* d_in, const int* in_sizes, int n_in,
                              void* d_out, int out_size, void* d_ws, size_t ws_size,
                              hipStream_t stream) {
  const int*   x   = (const int*)d_in[0];
  const float* emb = (const float*)d_in[1];
  const float* pos = (const float*)d_in[2];
  const float* Wq  = (const float*)d_in[3];
  const float* bq  = (const float*)d_in[4];
  const float* Wk  = (const float*)d_in[5];
  const float* bk  = (const float*)d_in[6];
  const float* Wv  = (const float*)d_in[7];
  const float* bv  = (const float*)d_in[8];
  const float* Wo  = (const float*)d_in[9];
  const float* bo  = (const float*)d_in[10];
  const float* W1  = (const float*)d_in[11];
  const float* b1  = (const float*)d_in[12];
  const float* W2  = (const float*)d_in[13];
  const float* b2  = (const float*)d_in[14];

  char* w = (char*)d_ws;
  auto alloc = [&](size_t bytes) -> void* {
    void* p = (void*)w;
    w += (bytes + 255) & ~(size_t)255;
    return p;
  };
  float*          hf  = (float*)alloc((size_t)Mm * Dd * 4);
  unsigned short* hb  = (unsigned short*)alloc((size_t)Mm * Dd * 2);
  unsigned short* qb  = (unsigned short*)alloc((size_t)Mm * HKk * 2);
  unsigned short* kb_ = (unsigned short*)alloc((size_t)Mm * HKk * 2);
  unsigned short* vb_ = (unsigned short*)alloc((size_t)Mm * HKk * 2);
  unsigned short* vtb = (unsigned short*)alloc((size_t)Mm * HKk * 2);
  unsigned short* ao  = (unsigned short*)alloc((size_t)Mm * HKk * 2);
  unsigned short* f1  = (unsigned short*)alloc((size_t)Mm * Ff * 2);
  unsigned short* WqT = (unsigned short*)alloc((size_t)Dd * HKk * 2);
  unsigned short* WkT = (unsigned short*)alloc((size_t)Dd * HKk * 2);
  unsigned short* WvT = (unsigned short*)alloc((size_t)Dd * HKk * 2);
  unsigned short* WoT = (unsigned short*)alloc((size_t)HKk * Dd * 2);
  unsigned short* W1T = (unsigned short*)alloc((size_t)Dd * Ff * 2);
  unsigned short* W2T = (unsigned short*)alloc((size_t)Ff * Dd * 2);

  const int TPB = 256;
  const int nH = Mm * Dd;
  const float qscale = 0.125f;  // 1/sqrt(64)

  embed_k<<<(nH + TPB - 1) / TPB, TPB, 0, stream>>>(x, emb, pos, hf, nH);
  conv_k<<<(nH + TPB - 1) / TPB, TPB, 0, stream>>>(hf, hb, nH);

  for (int l = 0; l < Ll; ++l) {
    const int nqkv = Dd * HKk, nff = Dd * Ff;
    trans_k<<<(nqkv + TPB - 1) / TPB, TPB, 0, stream>>>(Wq + (size_t)l * nqkv, WqT, Dd, HKk);
    trans_k<<<(nqkv + TPB - 1) / TPB, TPB, 0, stream>>>(Wk + (size_t)l * nqkv, WkT, Dd, HKk);
    trans_k<<<(nqkv + TPB - 1) / TPB, TPB, 0, stream>>>(Wv + (size_t)l * nqkv, WvT, Dd, HKk);
    trans_k<<<(nqkv + TPB - 1) / TPB, TPB, 0, stream>>>(Wo + (size_t)l * nqkv, WoT, HKk, Dd);
    trans_k<<<(nff + TPB - 1) / TPB, TPB, 0, stream>>>(W1 + (size_t)l * nff, W1T, Dd, Ff);
    trans_k<<<(nff + TPB - 1) / TPB, TPB, 0, stream>>>(W2 + (size_t)l * nff, W2T, Ff, Dd);

    // Q (pre-scaled), K, V projections (read hb)
    dim3 gqkv(HKk / 128, Mm / 128);
    gemm_k<0><<<gqkv, 256, 0, stream>>>(hb, WqT, bq + (size_t)l * HKk, qb, nullptr,
                                        Mm, HKk, Dd, qscale);
    gemm_k<0><<<gqkv, 256, 0, stream>>>(hb, WkT, bk + (size_t)l * HKk, kb_, nullptr,
                                        Mm, HKk, Dd, 1.0f);
    gemm_k<0><<<gqkv, 256, 0, stream>>>(hb, WvT, bv + (size_t)l * HKk, vb_, nullptr,
                                        Mm, HKk, Dd, 1.0f);

    // head-major V transpose (once per layer)
    vtrans_k<<<(nH + TPB - 1) / TPB, TPB, 0, stream>>>(vb_, vtb, nH);

    // attention
    attn_k<<<Bb * Hh * (Ss / 16), 32, 0, stream>>>(qb, kb_, vtb, ao);

    // output projection + residual into hf, bf16 copy into hb
    dim3 go(Dd / 128, Mm / 128);
    gemm_k<3><<<go, 256, 0, stream>>>(ao, WoT, bo + (size_t)l * Dd, hb, hf,
                                      Mm, Dd, HKk, 1.0f);

    // FFN: f1 = relu(hb @ W1 + b1); hf += f1 @ W2 + b2 (and refresh hb)
    dim3 g1(Ff / 128, Mm / 128);
    gemm_k<1><<<g1, 256, 0, stream>>>(hb, W1T, b1 + (size_t)l * Ff, f1, nullptr,
                                      Mm, Ff, Dd, 1.0f);
    gemm_k<3><<<go, 256, 0, stream>>>(f1, W2T, b2 + (size_t)l * Dd, hb, hf,
                                      Mm, Dd, Ff, 1.0f);
  }

  (void)hipMemcpyAsync(d_out, hf, (size_t)Mm * Dd * sizeof(float),
                       hipMemcpyDeviceToDevice, stream);
}
